// ConvFeature_84739704750242
// MI455X (gfx1250) — matrix operand order repaired and checked by reading, not compiled, not run on hardware
//
#include <hip/hip_runtime.h>
#include <hip/hip_bf16.h>

// ---------------------------------------------------------------------------
// ConvFeature (deformable multi-scale 1D conv) for MI455X / gfx1250.
// Dominant GEMM on v_wmma_f32_16x16x32_bf16; bf16 embedding table is
// L2-resident (256KB) so the deformable gathers never touch HBM; per-token
// offset predictor pre-tabulated.  Round-2 fix: software-pipelined WMMA loop
// + __launch_bounds__(512,1) to eliminate VGPR spills seen in round-1 asm.
// ---------------------------------------------------------------------------

typedef __attribute__((ext_vector_type(16))) __bf16 bf16x16;
typedef __attribute__((ext_vector_type(8)))  float  v8f;

#define WIDTH   512
#define GROUPS  2
#define WG_     256   // WIDTH / GROUPS
#define BB      16
#define SS      1024
#define VOCAB   256
#define EPS_    1e-5f

__device__ __forceinline__ float bf2f(unsigned short u) {
    return __uint_as_float(((unsigned int)u) << 16);
}
__device__ __forceinline__ unsigned short f2bf(float f) {
    unsigned int u = __float_as_uint(f);
    unsigned int r = u + 0x7FFFu + ((u >> 16) & 1u);  // round-to-nearest-even
    return (unsigned short)(r >> 16);
}

// -------------------------------------------------------------------- emb→bf16
__global__ void emb2bf_kernel(const float* __restrict__ emb,
                              unsigned short* __restrict__ out) {
    int i = blockIdx.x * blockDim.x + threadIdx.x;
    if (i < VOCAB * WIDTH) out[i] = f2bf(emb[i]);
}

// ------------------------------------------------------- per-token offset table
// offtab[v][k] = tanh(emb[v,:]·wo[:,k] + bo[k]) * OFFSET_SCALE
__global__ void offtab_kernel(const float* __restrict__ emb,
                              const float* __restrict__ wo,
                              const float* __restrict__ bo,
                              float* __restrict__ offtab, int K) {
    int w    = (blockIdx.x * blockDim.x + threadIdx.x) >> 5;  // wave id
    int lane = threadIdx.x & 31;
    if (w >= VOCAB * K) return;            // uniform per-wave, no divergence
    int v = w / K, k = w % K;
    float p = 0.f;
    for (int j = lane; j < WIDTH; j += 32) p += emb[v * WIDTH + j] * wo[j * K + k];
    #pragma unroll
    for (int d = 16; d > 0; d >>= 1) p += __shfl_down(p, d, 32);
    if (lane == 0) offtab[v * K + k] = tanhf(p + bo[k]) * 2.0f;
}

// -------------------------------------------- pack wc (f32) → bf16 B-fragments
// bpack layout: [g][kchunk=K*8][ntile=16][lane=32][e=16] halves.
// B 32x16 fragment: lanes 0-15 hold K=0..7,16..23 of column (lane&15),
//                   lanes 16-31 hold K=8..15,24..31.  kdim = tap*256 + i.
__global__ void pack_kernel(const float* __restrict__ wc,
                            unsigned short* __restrict__ bp, int K) {
    int idx = blockIdx.x * blockDim.x + threadIdx.x;
    int nk = K * 8;
    int total = GROUPS * nk * 16 * 512;
    if (idx >= total) return;
    int e   = idx & 15;
    int rem = idx >> 4;
    int lane = rem & 31; rem >>= 5;
    int nt   = rem & 15; rem >>= 4;
    int kc   = rem % nk;
    int g    = rem / nk;
    int c    = lane & 15;
    int kloc = (e < 8) ? (((lane >> 4) << 3) + e) : (16 + ((lane >> 4) << 3) + (e - 8));  // 0..31 within 32-chunk
    int o    = nt * 16 + c;
    int t    = kc >> 3;
    int i    = ((kc & 7) << 5) + kloc;          // input channel within group
    // wc layout [G][Wg][Wg][K] = (g, o, i, k)
    bp[idx] = f2bf(wc[(((g * WG_ + o) * WG_) + i) * K + t]);
}

// --------------------------------------------------- deformable conv via WMMA
// grid = (B*S/16 row-blocks, GROUPS); block = 512 threads = 16 waves.
// Per tap: stage interpolated 16x256 bf16 A-slab into LDS (A-fragment layout),
// then each wave runs a 2-deep software-pipelined chain of 8 WMMAs for its
// 16-column N tile.  __launch_bounds__(512,1): one WG per WGP -> 256 VGPR/wave
// budget, no spills.
template <int K>
__global__ void __launch_bounds__(512, 1)
conv_kernel(const int* __restrict__ tokens,
            const unsigned short* __restrict__ emb_bf,
            const float* __restrict__ offtab,
            const unsigned short* __restrict__ bpack,
            const float* __restrict__ bc,
            float* __restrict__ conv_out) {
    const int rb   = blockIdx.x;
    const int g    = blockIdx.y;
    const int tid  = threadIdx.x;
    const int lane = tid & 31;
    const int wave = tid >> 5;

    __shared__ alignas(32) unsigned short A_lds[8 * 512];  // one tap: 8 chunks x 512 halves
    __shared__ float sh_frac[16 * K];
    __shared__ int   sh_t0[16 * K];
    __shared__ int   sh_t1[16 * K];

    // ---- per (row, tap) sampling metadata: position -> token ids + frac ----
    if (tid < 16 * K) {
        int m = tid / K, t = tid % K;
        int row = rb * 16 + m;
        int b = row >> 10, s = row & (SS - 1);
        int tk = tokens[row];
        float off = offtab[tk * K + t];
        float pos = (float)s + ((float)t - (K - 1) * 0.5f) + off;
        float p0  = floorf(pos);
        int   i0  = (int)p0;
        sh_frac[tid] = pos - p0;
        sh_t0[tid] = (i0 >= 0 && i0 < SS) ? tokens[b * SS + i0] : -1;
        int i1 = i0 + 1;
        sh_t1[tid] = (i1 >= 0 && i1 < SS) ? tokens[b * SS + i1] : -1;
    }
    __syncthreads();

    v8f acc = {0.f, 0.f, 0.f, 0.f, 0.f, 0.f, 0.f, 0.f};
    const int nt = wave;  // 16 waves cover the 256 group-output channels

    for (int t = 0; t < K; ++t) {
        // ---- stage A slab for tap t (interp from L2-resident bf16 emb) ----
        for (int idx = tid; idx < 8 * 512; idx += 512) {
            int c8  = idx >> 9;           // 32-channel chunk within tap
            int rem = idx & 511;
            int l   = rem >> 4;
            int e   = rem & 15;
            int m   = l & 15;
            // A 16x32 bf16 fragment K-mapping (ISA 7.12.2)
            int kloc = (l < 16) ? ((e < 8) ? e : e + 8)
                                : ((e < 8) ? e + 8 : e + 16);
            int ch = (c8 << 5) + kloc;
            int mt = m * K + t;
            float f  = sh_frac[mt];
            int   t0 = sh_t0[mt], t1 = sh_t1[mt];
            int   cg = g * WG_ + ch;
            float v0 = (t0 >= 0) ? bf2f(emb_bf[t0 * WIDTH + cg]) : 0.f;
            float v1 = (t1 >= 0) ? bf2f(emb_bf[t1 * WIDTH + cg]) : 0.f;
            A_lds[idx] = f2bf(v0 + f * (v1 - v0));
        }
        __syncthreads();

        // ---- 8 WMMAs over this tap, 2-deep software pipeline -------------
        // Only (cur, next) fragment pairs live: ~40 VGPRs, no spills.
        const unsigned short* bptr =
            bpack + ((size_t)((g * K * 8 + t * 8) * 16 + nt) * 512) + lane * 16;
        const unsigned short* aptr = &A_lds[lane * 16];

        if (t + 1 < K)  // pull next tap's B slab toward L2/L0 while we compute
            __builtin_prefetch(bptr + (size_t)8 * (16 * 512), 0, 1);

        bf16x16 a_cur = *(const bf16x16*)(aptr);
        bf16x16 b_cur = *(const bf16x16*)(bptr);
        #pragma unroll
        for (int kc = 0; kc < 8; ++kc) {
            bf16x16 a_nxt, b_nxt;
            if (kc < 7) {
                a_nxt = *(const bf16x16*)(aptr + ((kc + 1) << 9));
                b_nxt = *(const bf16x16*)(bptr + (size_t)(kc + 1) * (16 * 512));
            }
            acc = __builtin_amdgcn_wmma_f32_16x16x32_bf16(
                false, a_cur, false, b_cur, (short)0, acc, false, false);
            a_cur = a_nxt;
            b_cur = b_nxt;
        }
        __syncthreads();
    }

    // ---- epilogue: C/D 16x16 layout (lane = N, VGPR r = M row) + bias ----
    int nl  = lane & 15;
    int col = g * WG_ + nt * 16 + nl;
    float bias  = bc[col];
    int   mbase = (lane < 16) ? 0 : 8;
    #pragma unroll
    for (int r = 0; r < 8; ++r) {
        int row = rb * 16 + mbase + r;
        conv_out[(size_t)row * WIDTH + col] = acc[r] + bias;
    }
}

// ------------------------------------- layernorm + mask + fold pw (per row)
__global__ void ln_kernel(const float* __restrict__ x,
                          const int* __restrict__ mask,
                          const float* __restrict__ g,
                          const float* __restrict__ b,
                          const float* __restrict__ pw,
                          float* __restrict__ contrib) {
    int row = blockIdx.x;
    int tid = threadIdx.x;  // 128
    const float* xr = x + (size_t)row * WIDTH;
    float v[4];
    float s = 0.f, sq = 0.f;
    #pragma unroll
    for (int i = 0; i < 4; ++i) {
        v[i] = xr[tid + i * 128];
        s += v[i];
        sq += v[i] * v[i];
    }
    __shared__ float s1[128], s2[128];
    s1[tid] = s; s2[tid] = sq;
    __syncthreads();
    for (int o = 64; o > 0; o >>= 1) {
        if (tid < o) { s1[tid] += s1[tid + o]; s2[tid] += s2[tid + o]; }
        __syncthreads();
    }
    float mu  = s1[0] * (1.f / WIDTH);
    float var = s2[0] * (1.f / WIDTH) - mu * mu;
    float rs  = rsqrtf(var + EPS_);
    __syncthreads();
    float p = 0.f;
    #pragma unroll
    for (int i = 0; i < 4; ++i) {
        int j = tid + i * 128;
        p += ((v[i] - mu) * rs * g[j] + b[j]) * pw[j];
    }
    s1[tid] = p;
    __syncthreads();
    for (int o = 64; o > 0; o >>= 1) {
        if (tid < o) s1[tid] += s1[tid + o];
        __syncthreads();
    }
    if (tid == 0) contrib[row] = mask[row] ? 0.f : s1[0];
}

// ---------------------------------------------- masked mean pool + pb → out
__global__ void final_kernel(const float* __restrict__ contrib,
                             const int* __restrict__ mask,
                             const float* __restrict__ pb,
                             float* __restrict__ out, int br) {
    int b = blockIdx.x, tid = threadIdx.x;  // 256
    float s = 0.f, c = 0.f;
    for (int i = tid; i < SS; i += 256) {
        s += contrib[b * SS + i];
        c += (mask[b * SS + i] == 0) ? 1.f : 0.f;
    }
    __shared__ float s1[256], s2[256];
    s1[tid] = s; s2[tid] = c;
    __syncthreads();
    for (int o = 128; o > 0; o >>= 1) {
        if (tid < o) { s1[tid] += s1[tid + o]; s2[tid] += s2[tid + o]; }
        __syncthreads();
    }
    if (tid == 0) out[b * 4 + br] = s1[0] / fmaxf(s2[0], 1.f) + pb[0];
}

// ---------------------------------------------------------------------------
extern "C" void kernel_launch(void* const* d_in, const int* in_sizes, int n_in,
                              void* d_out, int out_size, void* d_ws, size_t ws_size,
                              hipStream_t stream) {
    (void)in_sizes; (void)n_in; (void)out_size; (void)ws_size;

    const int*   tokens = (const int*)d_in[0];
    const int*   mask   = (const int*)d_in[1];
    const float* emb    = (const float*)d_in[2];
    float*       out    = (float*)d_out;

    // workspace layout (all 256B-aligned offsets)
    char* ws = (char*)d_ws;
    unsigned short* emb_bf  = (unsigned short*)(ws);                           // 262144 B
    float*          offtab  = (float*)(ws + 262144);                           // 9216 B (K<=9)
    unsigned short* bpack   = (unsigned short*)(ws + 262144 + 9216);           // 2359296 B (K<=9)
    float*          conv_ws = (float*)(ws + 262144 + 9216 + 2359296);          // 33554432 B
    float*          contrib = (float*)(ws + 262144 + 9216 + 2359296 + 33554432); // 65536 B

    emb2bf_kernel<<<(VOCAB * WIDTH + 511) / 512, 512, 0, stream>>>(emb, emb_bf);

    const int Ks[4] = {3, 5, 7, 9};
    for (int br = 0; br < 4; ++br) {
        const int K = Ks[br];
        const int base = 3 + br * 8;
        const float* wo = (const float*)d_in[base + 0];
        const float* bo = (const float*)d_in[base + 1];
        const float* wc = (const float*)d_in[base + 2];
        const float* bc = (const float*)d_in[base + 3];
        const float* lg = (const float*)d_in[base + 4];
        const float* lb = (const float*)d_in[base + 5];
        const float* pw = (const float*)d_in[base + 6];
        const float* pb = (const float*)d_in[base + 7];

        offtab_kernel<<<32 * K, 256, 0, stream>>>(emb, wo, bo, offtab, K);

        int total = GROUPS * K * 8 * 16 * 512;
        pack_kernel<<<(total + 255) / 256, 256, 0, stream>>>(wc, bpack, K);

        dim3 grid(BB * SS / 16, GROUPS);
        switch (K) {
            case 3: conv_kernel<3><<<grid, 512, 0, stream>>>(tokens, emb_bf, offtab, bpack, bc, conv_ws); break;
            case 5: conv_kernel<5><<<grid, 512, 0, stream>>>(tokens, emb_bf, offtab, bpack, bc, conv_ws); break;
            case 7: conv_kernel<7><<<grid, 512, 0, stream>>>(tokens, emb_bf, offtab, bpack, bc, conv_ws); break;
            case 9: conv_kernel<9><<<grid, 512, 0, stream>>>(tokens, emb_bf, offtab, bpack, bc, conv_ws); break;
        }

        ln_kernel<<<BB * SS, 128, 0, stream>>>(conv_ws, mask, lg, lb, pw, contrib);
        final_kernel<<<BB, 256, 0, stream>>>(contrib, mask, pb, out, br);
    }
}
